// Generator_Gumbel_softmax_89326729822822
// MI455X (gfx1250) — matrix operand order repaired and checked by reading, not compiled, not run on hardware
//
#include <hip/hip_runtime.h>
#include <hip/hip_bf16.h>
#include <math.h>

typedef __attribute__((ext_vector_type(16))) _Float16 v16h;
typedef __attribute__((ext_vector_type(8)))  _Float16 v8h;
typedef __attribute__((ext_vector_type(8)))  float    v8f;

#define TEMP_INV 1.25f          // 1 / 0.8
#define GEPS     1e-20f

// ---------------------------------------------------------------------------
// Prep kernels
// ---------------------------------------------------------------------------
__global__ void cvt_f16_kernel(const float* __restrict__ X, _Float16* __restrict__ Y, int total) {
    int i = blockIdx.x * blockDim.x + threadIdx.x;
    if (i < total) Y[i] = (_Float16)X[i];
}

// W is K x N (row major, f32). WT is Npad x K (row major, f16), zero-padded rows.
__global__ void transpose_cvt_kernel(const float* __restrict__ W, _Float16* __restrict__ WT,
                                     int K, int N, int Npad) {
    int idx = blockIdx.x * blockDim.x + threadIdx.x;
    int total = Npad * K;
    if (idx >= total) return;
    int n = idx / K;
    int k = idx - n * K;
    WT[idx] = (n < N) ? (_Float16)W[(size_t)k * N + n] : (_Float16)0.f;
}

__global__ void pad_bias_kernel(const float* __restrict__ b, float* __restrict__ bp,
                                int N, int Npad) {
    int i = blockIdx.x * blockDim.x + threadIdx.x;
    if (i < Npad) bp[i] = (i < N) ? b[i] : 0.f;
}

// ---------------------------------------------------------------------------
// WMMA GEMM:  C(M x N) = act( A(M x K, f16 row-major) * BT(N x K, f16)^T + bias )
// Block: 256 threads = 8 wave32, arranged 4(M) x 2(N); block tile 128 x 128.
// Wave tile 32 x 64: 2 A-fragments x 4 B-fragments -> 8 f32 accumulators.
// K, N are compile-time so all fragment strides fold into the 24-bit IOFFSET
// of global_load_b128 (two base pointers total, no per-iteration address VALU).
// ACT: 0 = ELU  -> f16 out, 1 = sigmoid -> f16 out, 2 = none -> f32 out.
// M % 128 == 0, N % 128 == 0, K % 128 == 0 (guaranteed by padding).
// ---------------------------------------------------------------------------
template <int ACT, int K, int N>
__global__ __launch_bounds__(256)
void gemm_wmma_kernel(const _Float16* __restrict__ A, const _Float16* __restrict__ BT,
                      const float* __restrict__ bias, void* __restrict__ Cout) {
    const int lane = threadIdx.x & 31;
    const int wave = threadIdx.x >> 5;        // 0..7
    const int wm   = wave >> 1;               // 0..3
    const int wn   = wave & 1;                // 0..1
    const int bm   = blockIdx.x * 128 + wm * 32;
    const int bn   = blockIdx.y * 128 + wn * 64;
    const int mlo  = lane & 15;               // A: row, B/C: column within 16-wide tile
    const int mhi  = lane >> 4;               // half-wave select

    // A fragment: lane holds row bm+mlo,
    //   halves 0..15 = K k + mhi*16 .. +16 (contiguous, matches B)
    const _Float16* Ab = A + (size_t)(bm + mlo) * K + mhi * 16;
    // B fragments from transposed weights: lane holds col bn + tj*16 + mlo,
    //   halves 0..15 = K k + mhi*16 .. +16 (contiguous in BT); tile stride 16*K (imm)
    const _Float16* Bb = BT + (size_t)(bn + mlo) * K + mhi * 16;

    v8f acc[2][4] = {};

    for (int k0 = 0; k0 < K; k0 += 128) {
        // Branch-free speculative prefetch of the streaming A operand two
        // outer steps ahead (OOB prefetch is silently dropped).
        __builtin_prefetch((const void*)(Ab + k0 + 256), 0, 3);
        __builtin_prefetch((const void*)(Ab + 16 * K + k0 + 256), 0, 3);

#pragma unroll
        for (int kk = 0; kk < 128; kk += 32) {
            const int k = k0 + kk;
            v8h a0lo = *(const v8h*)(Ab + k);
            v8h a0hi = *(const v8h*)(Ab + k + 8);
            v8h a1lo = *(const v8h*)(Ab + 16 * K + k);
            v8h a1hi = *(const v8h*)(Ab + 16 * K + k + 8);
            v16h a0, a1;
#pragma unroll
            for (int i = 0; i < 8; ++i) {
                a0[i] = a0lo[i]; a0[i + 8] = a0hi[i];
                a1[i] = a1lo[i]; a1[i + 8] = a1hi[i];
            }
            v16h b0 = *(const v16h*)(Bb + k);
            v16h b1 = *(const v16h*)(Bb + 16 * K + k);
            v16h b2 = *(const v16h*)(Bb + 32 * K + k);
            v16h b3 = *(const v16h*)(Bb + 48 * K + k);

            acc[0][0] = __builtin_amdgcn_wmma_f32_16x16x32_f16(false, a0, false, b0, (short)0, acc[0][0], false, false);
            acc[1][0] = __builtin_amdgcn_wmma_f32_16x16x32_f16(false, a1, false, b0, (short)0, acc[1][0], false, false);
            acc[0][1] = __builtin_amdgcn_wmma_f32_16x16x32_f16(false, a0, false, b1, (short)0, acc[0][1], false, false);
            acc[1][1] = __builtin_amdgcn_wmma_f32_16x16x32_f16(false, a1, false, b1, (short)0, acc[1][1], false, false);
            acc[0][2] = __builtin_amdgcn_wmma_f32_16x16x32_f16(false, a0, false, b2, (short)0, acc[0][2], false, false);
            acc[1][2] = __builtin_amdgcn_wmma_f32_16x16x32_f16(false, a1, false, b2, (short)0, acc[1][2], false, false);
            acc[0][3] = __builtin_amdgcn_wmma_f32_16x16x32_f16(false, a0, false, b3, (short)0, acc[0][3], false, false);
            acc[1][3] = __builtin_amdgcn_wmma_f32_16x16x32_f16(false, a1, false, b3, (short)0, acc[1][3], false, false);
        }
    }

    // Epilogue. C/D layout: lane column = mlo (within tile), rows = r + mhi*8.
#pragma unroll
    for (int ti = 0; ti < 2; ++ti) {
#pragma unroll
        for (int tj = 0; tj < 4; ++tj) {
            const int col  = bn + tj * 16 + mlo;
            const int row0 = bm + ti * 16 + mhi * 8;
            const float bc = bias[col];
#pragma unroll
            for (int r = 0; r < 8; ++r) {
                float v = acc[ti][tj][r] + bc;
                if (ACT == 0) v = v > 0.f ? v : (__expf(v) - 1.f);          // ELU
                else if (ACT == 1) v = 1.f / (1.f + __expf(-v));            // sigmoid
                if (ACT == 2)
                    ((float*)Cout)[(size_t)(row0 + r) * N + col] = v;
                else
                    ((_Float16*)Cout)[(size_t)(row0 + r) * N + col] = (_Float16)v;
            }
        }
    }
}

// ---------------------------------------------------------------------------
// Segmented Gumbel softmax: one wave32 per row. 8 rows per 256-thread block.
// ---------------------------------------------------------------------------
__global__ __launch_bounds__(256)
void gumbel_softmax_kernel(const float* __restrict__ logits, const float* __restrict__ u,
                           float* __restrict__ out, int ldl, int B) {
    const int NSEG = 13;
    const int seg_n[NSEG]   = {29, 2, 1, 6, 7, 59, 1, 24, 76, 711, 3, 9, 1};
    const int seg_off[NSEG] = {0, 29, 31, 32, 38, 45, 104, 105, 129, 205, 916, 919, 928};

    const int wave = threadIdx.x >> 5;
    const int lane = threadIdx.x & 31;
    const int row  = blockIdx.x * 8 + wave;
    if (row >= B) return;

    const float* L = logits + (size_t)row * ldl;
    const float* U = u      + (size_t)row * 929;
    float*       O = out    + (size_t)row * 929;

    for (int s = 0; s < NSEG; ++s) {
        const int n = seg_n[s], off = seg_off[s];
        if (n == 1) {
            if (lane == 0) O[off] = L[off];
            continue;
        }
        // pass 1: max of y = (logit + gumbel) / T
        float mx = -3.4e38f;
        for (int i = lane; i < n; i += 32) {
            float g = -__logf(-__logf(U[off + i] + GEPS) + GEPS);
            float y = (L[off + i] + g) * TEMP_INV;
            mx = fmaxf(mx, y);
        }
#pragma unroll
        for (int o = 16; o >= 1; o >>= 1) mx = fmaxf(mx, __shfl_xor(mx, o, 32));
        // pass 2: exp and sum
        float sum = 0.f;
        for (int i = lane; i < n; i += 32) {
            float g = -__logf(-__logf(U[off + i] + GEPS) + GEPS);
            float y = (L[off + i] + g) * TEMP_INV;
            float e = __expf(y - mx);
            O[off + i] = e;
            sum += e;
        }
#pragma unroll
        for (int o = 16; o >= 1; o >>= 1) sum += __shfl_xor(sum, o, 32);
        // pass 3: normalize
        const float inv = 1.f / sum;
        for (int i = lane; i < n; i += 32) O[off + i] *= inv;
    }
}

// ---------------------------------------------------------------------------
// Host launch
// ---------------------------------------------------------------------------
extern "C" void kernel_launch(void* const* d_in, const int* in_sizes, int n_in,
                              void* d_out, int out_size, void* d_ws, size_t ws_size,
                              hipStream_t stream) {
    const float* x  = (const float*)d_in[0];
    const float* W1 = (const float*)d_in[1];
    const float* b1 = (const float*)d_in[2];
    const float* W2 = (const float*)d_in[3];
    const float* b2 = (const float*)d_in[4];
    const float* W3 = (const float*)d_in[5];
    const float* b3 = (const float*)d_in[6];
    const float* u  = (const float*)d_in[7];

    const int B = 32768, D = 128, H = 1024, O = 929, OP = 1024; // OP: O padded to 128-multiple

    char* ws = (char*)d_ws;
    size_t off = 0;
    auto alloc = [&](size_t bytes) -> void* {
        void* p = ws + off;
        off = (off + bytes + 255) & ~(size_t)255;
        return p;
    };
    _Float16* x16 = (_Float16*)alloc((size_t)B * D * 2);
    _Float16* W1T = (_Float16*)alloc((size_t)H * D * 2);
    _Float16* W2T = (_Float16*)alloc((size_t)H * H * 2);
    _Float16* W3T = (_Float16*)alloc((size_t)OP * H * 2);
    float*    b3p = (float*)alloc((size_t)OP * 4);
    _Float16* h1  = (_Float16*)alloc((size_t)B * H * 2);
    _Float16* h2  = (_Float16*)alloc((size_t)B * H * 2);
    float*    lg  = (float*)alloc((size_t)B * OP * 4);

    // prep
    cvt_f16_kernel<<<(B * D + 255) / 256, 256, 0, stream>>>(x, x16, B * D);
    transpose_cvt_kernel<<<(H * D + 255) / 256, 256, 0, stream>>>(W1, W1T, D, H, H);
    transpose_cvt_kernel<<<(H * H + 255) / 256, 256, 0, stream>>>(W2, W2T, H, H, H);
    transpose_cvt_kernel<<<(OP * H + 255) / 256, 256, 0, stream>>>(W3, W3T, H, O, OP);
    pad_bias_kernel<<<(OP + 255) / 256, 256, 0, stream>>>(b3, b3p, O, OP);

    // 3 WMMA GEMMs with fused epilogues (all N padded to 1024)
    gemm_wmma_kernel<0, 128,  1024><<<dim3(B / 128, H  / 128), 256, 0, stream>>>(x16, W1T, b1, h1);
    gemm_wmma_kernel<1, 1024, 1024><<<dim3(B / 128, H  / 128), 256, 0, stream>>>(h1, W2T, b2, h2);
    gemm_wmma_kernel<2, 1024, 1024><<<dim3(B / 128, OP / 128), 256, 0, stream>>>(h2, W3T, b3p, lg);

    // segmented gumbel softmax -> d_out (32768 x 929 f32)
    gumbel_softmax_kernel<<<B / 8, 256, 0, stream>>>(lg, u, (float*)d_out, OP, B);
}
